// CrossAttentionLayer_45543833207141
// MI455X (gfx1250) — matrix operand order repaired and checked by reading, not compiled, not run on hardware
//
#include <hip/hip_runtime.h>
#include <hip/hip_bf16.h>

// ---------------------------------------------------------------------------
// CrossAttentionLayer for MI455X (gfx1250, wave32, WMMA, TDM)
//   Q = q@Wq^T+bq ; K = k@Wk^T+bk ; V = v@Wv^T+bv       (f16 WMMA GEMMs)
//   ctx = softmax(QK^T/8 + mask) @ V                     (flash, f16 WMMA)
//   out = LN(ctx@Wo^T + bo + q)                          (fused WMMA GEMM+LN)
// gemm_proj and flash_attn are software-pipelined: global loads for tile i+1 issue
// before the compute of tile i, LDS double-buffered, one barrier/iteration.
// ---------------------------------------------------------------------------

typedef _Float16 f16;
typedef __attribute__((ext_vector_type(4)))  _Float16 v4h;
typedef __attribute__((ext_vector_type(8)))  _Float16 v8h;
typedef __attribute__((ext_vector_type(16))) _Float16 v16h;
typedef __attribute__((ext_vector_type(8)))  float    v8f;
typedef unsigned int u32x4 __attribute__((ext_vector_type(4)));
typedef int          i32x4 __attribute__((ext_vector_type(4)));
typedef int          i32x8 __attribute__((ext_vector_type(8)));

#define D_MODEL 512
#define N_HEADS 8
#define D_K     64
#define BATCH   4
#define SEQ     2048
#define M_TOT   (BATCH * SEQ)   // 8192 rows

#if defined(__gfx1250__) && __has_builtin(__builtin_amdgcn_tensor_load_to_lds)
#define USE_TDM 1
#else
#define USE_TDM 0
#endif

__device__ __forceinline__ v8f wmma16(v16h a, v16h b, v8f c) {
  return __builtin_amdgcn_wmma_f32_16x16x32_f16(false, a, false, b,
                                                (short)0, c, false, false);
}

// A-fragment (16x32 f16): lane l -> row (l&15), elements 0..7 = K (l>>4)*8..+7,
// elements 8..15 = K (l>>4)*8+16..+23 (ISA 7.12.2).
__device__ __forceinline__ v16h fragA(const f16* rowPtr, int lane) {
  const int kb = (lane >> 4) * 8;
  v8h lo = *(const v8h*)(rowPtr + kb);
  v8h hi = *(const v8h*)(rowPtr + kb + 16);
  v16h r;
#pragma unroll
  for (int i = 0; i < 8; ++i) { r[i] = lo[i]; r[i + 8] = hi[i]; }
  return r;
}

// B-fragment (32x16 f16): lane l -> column (l&15), elements 0..7 = K (l>>4)*8..+7,
// elements 8..15 = K (l>>4)*8+16..+23.  colPtr -> K-contiguous column data.
__device__ __forceinline__ v16h fragB(const f16* colPtr, int lane) {
  const int kb = (lane >> 4) * 8;
  v8h lo = *(const v8h*)(colPtr + kb);
  v8h hi = *(const v8h*)(colPtr + kb + 16);
  v16h r;
#pragma unroll
  for (int i = 0; i < 8; ++i) { r[i] = lo[i]; r[i + 8] = hi[i]; }
  return r;
}

// Cross-lane reductions inside a 16-lane half (ds_swizzle group-of-32 XOR).
#define SWZ_XOR_F(v, imm) \
  __int_as_float(__builtin_amdgcn_ds_swizzle(__float_as_int(v), (((imm) << 10) | 0x1f)))

__device__ __forceinline__ float half16_max(float v) {
  v = fmaxf(v, SWZ_XOR_F(v, 1));
  v = fmaxf(v, SWZ_XOR_F(v, 2));
  v = fmaxf(v, SWZ_XOR_F(v, 4));
  v = fmaxf(v, SWZ_XOR_F(v, 8));
  return v;
}
__device__ __forceinline__ float half16_sum(float v) {
  v += SWZ_XOR_F(v, 1);
  v += SWZ_XOR_F(v, 2);
  v += SWZ_XOR_F(v, 4);
  v += SWZ_XOR_F(v, 8);
  return v;
}

// ---------------------------------------------------------------------------
// Kernel 0: fp32 -> f16 weight conversion (2 MB total; L2-resident after).
// ---------------------------------------------------------------------------
__global__ __launch_bounds__(256) void cvt_f32_f16(const float* __restrict__ src,
                                                   f16* __restrict__ dst, int n) {
  int i = (blockIdx.x * 256 + threadIdx.x) * 4;
  if (i + 3 < n) {
    float4 f = *(const float4*)(src + i);
    v4h o;
    o[0] = (f16)f.x; o[1] = (f16)f.y; o[2] = (f16)f.z; o[3] = (f16)f.w;
    *(v4h*)(dst + i) = o;
  }
}

// ---------------------------------------------------------------------------
// Kernel 1: Y(f16)[M,512] = X(f32)[M,512] @ W16^T + bias.  Tile 128x64,
// 8 waves; software-pipelined, double-buffered LDS, one barrier/K-step.
// ---------------------------------------------------------------------------
__global__ __launch_bounds__(256) void gemm_proj(const float* __restrict__ X,
                                                 const f16* __restrict__ W16,
                                                 const float* __restrict__ bias,
                                                 f16* __restrict__ Y) {
  __shared__ __align__(16) f16 Xs[2][128 * 40];
  __shared__ __align__(16) f16 Ws[2][64 * 40];
  const int tid = threadIdx.x, lane = tid & 31, wv = tid >> 5;
  const int ln = lane & 15, hl = lane >> 4;
  const int mBase = blockIdx.x * 128, nBase = blockIdx.y * 64;

  // staging coordinates
  const int xr_ = tid >> 1, xc_ = (tid & 1) * 16;   // X: 16 fp32/thread
  const int wr_ = tid >> 2, wc_ = (tid & 3) * 8;    // W: 8 f16/thread
  const float* Xrow = X + (size_t)(mBase + xr_) * D_MODEL + xc_;
  const f16*   Wrow = W16 + (size_t)(nBase + wr_) * D_MODEL + wc_;

  v8f acc[4] = {};

  // prologue: stage K-step 0 into buffer 0
  {
#pragma unroll
    for (int i = 0; i < 16; i += 4) {
      float4 f = *(const float4*)(Xrow + i);
      v4h o; o[0] = (f16)f.x; o[1] = (f16)f.y; o[2] = (f16)f.z; o[3] = (f16)f.w;
      *(v4h*)(&Xs[0][xr_ * 40 + xc_] + i) = o;
    }
    *(v8h*)&Ws[0][wr_ * 40 + wc_] = *(const v8h*)(Wrow);
  }

  for (int it = 0; it < D_MODEL / 32; ++it) {
    const int cur = it & 1, nxt = cur ^ 1;
    const bool has = (it + 1 < D_MODEL / 32);
    __syncthreads();

    // issue next tile's global loads (hidden under the WMMA work below)
    float4 f0, f1, f2, f3; v8h wreg;
    if (has) {
      const float* s = Xrow + (it + 1) * 32;
      f0 = *(const float4*)(s + 0);
      f1 = *(const float4*)(s + 4);
      f2 = *(const float4*)(s + 8);
      f3 = *(const float4*)(s + 12);
      wreg = *(const v8h*)(Wrow + (it + 1) * 32);
    }

    // compute current tile: one ds clause, then 4 back-to-back WMMAs
    v16h a  = fragA(&Xs[cur][(wv * 16 + ln) * 40], lane);
    v16h b0 = fragB(&Ws[cur][(0 * 16 + ln) * 40], lane);
    v16h b1 = fragB(&Ws[cur][(1 * 16 + ln) * 40], lane);
    v16h b2 = fragB(&Ws[cur][(2 * 16 + ln) * 40], lane);
    v16h b3 = fragB(&Ws[cur][(3 * 16 + ln) * 40], lane);
    acc[0] = wmma16(a, b0, acc[0]);
    acc[1] = wmma16(a, b1, acc[1]);
    acc[2] = wmma16(a, b2, acc[2]);
    acc[3] = wmma16(a, b3, acc[3]);

    // commit staged registers to the other buffer
    if (has) {
      f16* d = &Xs[nxt][xr_ * 40 + xc_];
      v4h o;
      o[0] = (f16)f0.x; o[1] = (f16)f0.y; o[2] = (f16)f0.z; o[3] = (f16)f0.w;
      *(v4h*)(d + 0) = o;
      o[0] = (f16)f1.x; o[1] = (f16)f1.y; o[2] = (f16)f1.z; o[3] = (f16)f1.w;
      *(v4h*)(d + 4) = o;
      o[0] = (f16)f2.x; o[1] = (f16)f2.y; o[2] = (f16)f2.z; o[3] = (f16)f2.w;
      *(v4h*)(d + 8) = o;
      o[0] = (f16)f3.x; o[1] = (f16)f3.y; o[2] = (f16)f3.z; o[3] = (f16)f3.w;
      *(v4h*)(d + 12) = o;
      *(v8h*)&Ws[nxt][wr_ * 40 + wc_] = wreg;
    }
  }

#pragma unroll
  for (int j = 0; j < 4; ++j) {
    const int col = nBase + j * 16 + ln;
    const float bv = bias[col];
#pragma unroll
    for (int i = 0; i < 8; ++i) {
      const int row = mBase + wv * 16 + i + 8 * hl;
      Y[(size_t)row * D_MODEL + col] = (f16)(acc[j][i] + bv);
    }
  }
}

// ---------------------------------------------------------------------------
// Kernel 2: flash attention.  Block = (b,h, 128 q-rows); wave owns 16 q-rows.
// K-tile staged by the Tensor Data Mover (32x64 f16 tile, stride-512 tensor,
// LDS row padding 128B+16B via pad_interval/pad_amount), V-tile transposed
// through registers; both double-buffered, one barrier per 32-key step.
// ---------------------------------------------------------------------------
#if USE_TDM
typedef __attribute__((address_space(3))) f16 lds_f16;
__device__ __forceinline__ unsigned lds_offset(f16* p) {
  return (unsigned)(unsigned long long)(lds_f16*)p;
}
#endif

__global__ __launch_bounds__(256) void flash_attn(const f16* __restrict__ Q,
                                                  const f16* __restrict__ K,
                                                  const f16* __restrict__ V,
                                                  const unsigned char* __restrict__ mask,
                                                  f16* __restrict__ CTX) {
  __shared__ __align__(16) f16 Ks[2][32 * 72];     // 32 keys x 64 dk (+8 pad)
  __shared__ __align__(16) f16 Vt[2][64 * 40];     // transposed: 64 dk x 32 keys
  __shared__ __align__(16) f16 Ps[8][16 * 40];     // per-wave P (16 x 32)

  const int tid = threadIdx.x, lane = tid & 31, wv = tid >> 5;
  const int ln = lane & 15, hl = lane >> 4;
  const int h = blockIdx.y, b = blockIdx.z;
  const int qBase = blockIdx.x * 128;
  const size_t rowOff = (size_t)b * SEQ;
  const float scale = 0.125f;  // 1/sqrt(64)

  const f16* Kbase = K + rowOff * D_MODEL + h * D_K;
  const f16* Vbase = V + rowOff * D_MODEL + h * D_K;
  const int sr = tid >> 3;         // staging key row 0..31
  const int sc = (tid & 7) * 8;    // staging dk col {0,8,..,56}

  // Q fragments for this wave's 16 rows (two K-halves of d_k = 64)
  const f16* qp = Q + (rowOff + qBase + wv * 16 + ln) * D_MODEL + h * D_K;
  const v16h aq0 = fragA(qp, lane);
  const v16h aq1 = fragA(qp + 32, lane);

  float mrow[8], lrow[8];
#pragma unroll
  for (int i = 0; i < 8; ++i) { mrow[i] = -1e30f; lrow[i] = 0.0f; }
  v8f o[4] = {};

#if USE_TDM
  // D# group 1 (invariant): data_size=2B, pad every 32 DWORDs by 4 DWORDs,
  // tensor 64 x 32 (from tile origin), tile 64 x 32, dim0 stride 512.
  i32x8 g1;
  g1[0] = (1 << 16) | (1 << 20) | (4 << 22) | (3 << 25);
  g1[1] = (int)(64u << 16);          // tensor_dim0 low16 @ bits[63:48]
  g1[2] = (int)(32u << 16);          // tensor_dim1 low16 @ bits[95:80]
  g1[3] = (int)(64u << 16);          // tile_dim0 @ bits[127:112]
  g1[4] = 32;                        // tile_dim1 @ bits[143:128]
  g1[5] = D_MODEL;                   // tensor_dim0_stride low32
  g1[6] = 0;
  g1[7] = 0;
  const i32x4 gz4 = {};
  const i32x8 gz8 = {};
  const unsigned ldsK[2] = { lds_offset(&Ks[0][0]), lds_offset(&Ks[1][0]) };
#endif

  // ---- prologue: stage key block 0 into buffer 0
  {
    v8h vreg = *(const v8h*)(Vbase + (size_t)sr * D_MODEL + sc);
#pragma unroll
    for (int i = 0; i < 8; ++i) Vt[0][(sc + i) * 40 + sr] = vreg[i];
#if USE_TDM
    if (wv == 0) {
      unsigned long long ga = (unsigned long long)Kbase;
      u32x4 g0;
      g0[0] = 1u;                                  // count=1
      g0[1] = ldsK[0];                             // lds_addr
      g0[2] = (unsigned)ga;                        // global_addr lo
      g0[3] = (unsigned)(ga >> 32) | (2u << 30);   // global_addr hi | type=2
      __builtin_amdgcn_tensor_load_to_lds(g0, g1, gz4, gz4, gz8, 0);
      __builtin_amdgcn_s_wait_tensorcnt(0);
    }
#else
    *(v8h*)&Ks[0][sr * 72 + sc] =
        *(const v8h*)(Kbase + (size_t)sr * D_MODEL + sc);
#endif
  }

  for (int it = 0; it < SEQ / 32; ++it) {
    const int cur = it & 1, nxt = cur ^ 1;
    const int kb0 = it * 32;
    const bool has = (it + 1 < SEQ / 32);
    __syncthreads();

    // issue next block's loads (overlap with compute below)
    v8h vreg;
    if (has) {
      vreg = *(const v8h*)(Vbase + (size_t)(kb0 + 32 + sr) * D_MODEL + sc);
#if USE_TDM
      if (wv == 0) {
        unsigned long long ga =
            (unsigned long long)(Kbase + (size_t)(kb0 + 32) * D_MODEL);
        u32x4 g0;
        g0[0] = 1u;
        g0[1] = ldsK[nxt];
        g0[2] = (unsigned)ga;
        g0[3] = (unsigned)(ga >> 32) | (2u << 30);
        __builtin_amdgcn_tensor_load_to_lds(g0, g1, gz4, gz4, gz8, 0);
      }
#endif
    }
#if !USE_TDM
    v8h kreg;
    if (has)
      kreg = *(const v8h*)(Kbase + (size_t)(kb0 + 32 + sr) * D_MODEL + sc);
#endif

    // ---- S = Q * K^T : batch all B-fragments, then 4 WMMAs back-to-back
    v16h b00 = fragB(&Ks[cur][ln * 72], lane);
    v16h b01 = fragB(&Ks[cur][ln * 72 + 32], lane);
    v16h b10 = fragB(&Ks[cur][(16 + ln) * 72], lane);
    v16h b11 = fragB(&Ks[cur][(16 + ln) * 72 + 32], lane);
    v8f s0 = {}, s1 = {};
    s0 = wmma16(aq0, b00, s0);
    s0 = wmma16(aq1, b01, s0);
    s1 = wmma16(aq0, b10, s1);
    s1 = wmma16(aq1, b11, s1);

    const float mk0 = mask[(size_t)b * SEQ + kb0 + ln] ? 0.0f : -1e9f;
    const float mk1 = mask[(size_t)b * SEQ + kb0 + 16 + ln] ? 0.0f : -1e9f;

    // ---- online softmax (rows live in 16-lane halves)
    f16* pw = &Ps[wv][0];
#pragma unroll
    for (int i = 0; i < 8; ++i) {
      const float a0 = s0[i] * scale + mk0;
      const float a1 = s1[i] * scale + mk1;
      const float mt = half16_max(fmaxf(a0, a1));
      const float mn = fmaxf(mrow[i], mt);
      const float corr = __expf(mrow[i] - mn);
      const float p0 = __expf(a0 - mn);
      const float p1 = __expf(a1 - mn);
      lrow[i] = lrow[i] * corr + half16_sum(p0 + p1);
      mrow[i] = mn;
#pragma unroll
      for (int j = 0; j < 4; ++j) o[j][i] *= corr;
      pw[(i + 8 * hl) * 40 + ln] = (f16)p0;         // C-layout -> row-major P
      pw[(i + 8 * hl) * 40 + 16 + ln] = (f16)p1;
    }

    // ---- O += P(16x32) * V(32x64): batch loads, then 4 WMMAs
    v16h ap  = fragA(&Ps[wv][ln * 40], lane);
    v16h bv0 = fragB(&Vt[cur][(0 * 16 + ln) * 40], lane);
    v16h bv1 = fragB(&Vt[cur][(1 * 16 + ln) * 40], lane);
    v16h bv2 = fragB(&Vt[cur][(2 * 16 + ln) * 40], lane);
    v16h bv3 = fragB(&Vt[cur][(3 * 16 + ln) * 40], lane);
    o[0] = wmma16(ap, bv0, o[0]);
    o[1] = wmma16(ap, bv1, o[1]);
    o[2] = wmma16(ap, bv2, o[2]);
    o[3] = wmma16(ap, bv3, o[3]);

    // ---- commit next V tile (transposed) and drain next K TDM
    if (has) {
#pragma unroll
      for (int i = 0; i < 8; ++i) Vt[nxt][(sc + i) * 40 + sr] = vreg[i];
#if USE_TDM
      if (wv == 0) __builtin_amdgcn_s_wait_tensorcnt(0);
#else
      *(v8h*)&Ks[nxt][sr * 72 + sc] = kreg;
#endif
    }
  }

#pragma unroll
  for (int i = 0; i < 8; ++i) {
    const float inv = 1.0f / lrow[i];
    const int row = qBase + wv * 16 + i + 8 * hl;
#pragma unroll
    for (int j = 0; j < 4; ++j) {
      CTX[(rowOff + row) * D_MODEL + h * D_K + j * 16 + ln] = (f16)(o[j][i] * inv);
    }
  }
}

// ---------------------------------------------------------------------------
// Kernel 3: out = LayerNorm(ctx @ Wo^T + bo + query).  Block owns 16 full
// rows (N = 512 over 8 waves) so LayerNorm stays block-local.
// ---------------------------------------------------------------------------
__global__ __launch_bounds__(256) void outproj_ln(const f16* __restrict__ CTX,
                                                  const f16* __restrict__ Wo,
                                                  const float* __restrict__ bo,
                                                  const float* __restrict__ resid,
                                                  const float* __restrict__ gamma,
                                                  const float* __restrict__ beta,
                                                  float* __restrict__ out) {
  __shared__ __align__(16) f16 As[16 * 40];
  __shared__ float rSum[16], rSqs[16];
  const int tid = threadIdx.x, lane = tid & 31, wv = tid >> 5;
  const int ln = lane & 15, hl = lane >> 4;
  const int mBase = blockIdx.x * 16;

  if (tid < 16) { rSum[tid] = 0.0f; rSqs[tid] = 0.0f; }
  v8f acc[4] = {};

  for (int k0 = 0; k0 < D_MODEL; k0 += 32) {
    __syncthreads();
    {
      const int r = tid >> 4, kc = (tid & 15) * 2;
      const f16* s = CTX + (size_t)(mBase + r) * D_MODEL + k0 + kc;
      As[r * 40 + kc] = s[0];
      As[r * 40 + kc + 1] = s[1];
    }
    __syncthreads();
    v16h a = fragA(&As[ln * 40], lane);
    v16h b0 = fragB(Wo + (size_t)(wv * 64 + 0 * 16 + ln) * D_MODEL + k0, lane);
    v16h b1 = fragB(Wo + (size_t)(wv * 64 + 1 * 16 + ln) * D_MODEL + k0, lane);
    v16h b2 = fragB(Wo + (size_t)(wv * 64 + 2 * 16 + ln) * D_MODEL + k0, lane);
    v16h b3 = fragB(Wo + (size_t)(wv * 64 + 3 * 16 + ln) * D_MODEL + k0, lane);
    acc[0] = wmma16(a, b0, acc[0]);
    acc[1] = wmma16(a, b1, acc[1]);
    acc[2] = wmma16(a, b2, acc[2]);
    acc[3] = wmma16(a, b3, acc[3]);
  }

#pragma unroll
  for (int j = 0; j < 4; ++j) {
    const int col = wv * 64 + j * 16 + ln;
    const float bb = bo[col];
#pragma unroll
    for (int i = 0; i < 8; ++i) {
      const int row = mBase + i + 8 * hl;
      acc[j][i] += bb + resid[(size_t)row * D_MODEL + col];
    }
  }

#pragma unroll
  for (int i = 0; i < 8; ++i) {
    float s = acc[0][i] + acc[1][i] + acc[2][i] + acc[3][i];
    float q = acc[0][i] * acc[0][i] + acc[1][i] * acc[1][i] +
              acc[2][i] * acc[2][i] + acc[3][i] * acc[3][i];
    s = half16_sum(s);
    q = half16_sum(q);
    if (ln == 0) {
      atomicAdd(&rSum[i + 8 * hl], s);
      atomicAdd(&rSqs[i + 8 * hl], q);
    }
  }
  __syncthreads();

#pragma unroll
  for (int i = 0; i < 8; ++i) {
    const int row = i + 8 * hl;
    const float mu = rSum[row] * (1.0f / D_MODEL);
    const float var = rSqs[row] * (1.0f / D_MODEL) - mu * mu;
    const float rstd = rsqrtf(var + 1e-5f);
#pragma unroll
    for (int j = 0; j < 4; ++j) {
      const int col = wv * 64 + j * 16 + ln;
      out[(size_t)(mBase + row) * D_MODEL + col] =
          (acc[j][i] - mu) * rstd * gamma[col] + beta[col];
    }
  }
}

// ---------------------------------------------------------------------------
extern "C" void kernel_launch(void* const* d_in, const int* in_sizes, int n_in,
                              void* d_out, int out_size, void* d_ws, size_t ws_size,
                              hipStream_t stream) {
  (void)in_sizes; (void)n_in; (void)out_size; (void)ws_size;
  const float* query = (const float*)d_in[0];
  const float* key   = (const float*)d_in[1];
  const float* value = (const float*)d_in[2];
  const unsigned char* kmask = (const unsigned char*)d_in[3];
  const float* Wq = (const float*)d_in[4];
  const float* bq = (const float*)d_in[5];
  const float* Wk = (const float*)d_in[6];
  const float* bk = (const float*)d_in[7];
  const float* Wv = (const float*)d_in[8];
  const float* bv = (const float*)d_in[9];
  const float* Wo = (const float*)d_in[10];
  const float* bo = (const float*)d_in[11];
  const float* gamma = (const float*)d_in[12];
  const float* beta  = (const float*)d_in[13];
  float* out = (float*)d_out;

  char* ws = (char*)d_ws;
  const size_t WSZ = (size_t)D_MODEL * D_MODEL * sizeof(f16);  // 512 KB
  const size_t ASZ = (size_t)M_TOT * D_MODEL * sizeof(f16);    // 8 MB
  f16* Wq16 = (f16*)(ws + 0 * WSZ);
  f16* Wk16 = (f16*)(ws + 1 * WSZ);
  f16* Wv16 = (f16*)(ws + 2 * WSZ);
  f16* Wo16 = (f16*)(ws + 3 * WSZ);
  f16* Q16 = (f16*)(ws + 4 * WSZ + 0 * ASZ);
  f16* K16 = (f16*)(ws + 4 * WSZ + 1 * ASZ);
  f16* V16 = (f16*)(ws + 4 * WSZ + 2 * ASZ);
  f16* C16 = (f16*)(ws + 4 * WSZ + 3 * ASZ);

  {
    const int n = D_MODEL * D_MODEL;
    dim3 g(n / (4 * 256));
    cvt_f32_f16<<<g, 256, 0, stream>>>(Wq, Wq16, n);
    cvt_f32_f16<<<g, 256, 0, stream>>>(Wk, Wk16, n);
    cvt_f32_f16<<<g, 256, 0, stream>>>(Wv, Wv16, n);
    cvt_f32_f16<<<g, 256, 0, stream>>>(Wo, Wo16, n);
  }
  dim3 gp(M_TOT / 128, D_MODEL / 64);
  gemm_proj<<<gp, 256, 0, stream>>>(query, Wq16, bq, Q16);
  gemm_proj<<<gp, 256, 0, stream>>>(key,   Wk16, bk, K16);
  gemm_proj<<<gp, 256, 0, stream>>>(value, Wv16, bv, V16);

  dim3 ga(SEQ / 128, N_HEADS, BATCH);
  flash_attn<<<ga, 256, 0, stream>>>(Q16, K16, V16, kmask, C16);

  outproj_ln<<<M_TOT / 16, 256, 0, stream>>>(C16, Wo16, bo, query, gamma, beta, out);
}
